// MAMoE_90512140795912
// MI455X (gfx1250) — compile-verified, matrix operand order checked by reading, not run on hardware
//
#include <hip/hip_runtime.h>

typedef __bf16 bf16;
typedef __attribute__((ext_vector_type(16))) __bf16 v16bf;
typedef __attribute__((ext_vector_type(8)))  float  v8f;

#define N_TOK 36864
#define SCALE_F 0.15430334996209191f  // (128/3)^-0.5 = 42^-0.5

// ---------------- WMMA helpers ----------------

__device__ inline v8f wmma_bf16(v16bf a, v16bf b, v8f c) {
    // D = A(16x32) * B(32x16) + C(16x16), f32 accum
    return __builtin_amdgcn_wmma_f32_16x16x32_bf16(
        false, a, false, b, (short)0, c, false, false);
}

// Fragment for A row / B column when contiguous in k.
// Caller passes p = rowBase (already includes (lane&15) row select and k-chunk base).
__device__ inline v16bf frag_row(const bf16* p, int lane) {
    const bf16* q = p + ((lane >> 4) << 3);
    v16bf f;
#pragma unroll
    for (int i = 0; i < 8; ++i) { f[i] = q[i]; f[i + 8] = q[i + 16]; }
    return f;
}

// Gather a B column fragment when B is row-major with leading dim ld.
__device__ inline v16bf frag_col_gather(const bf16* base, int ld, int ncol, int kb, int lane) {
    int kh = (lane >> 4) << 3;
    v16bf f;
#pragma unroll
    for (int i = 0; i < 8; ++i) {
        f[i]     = base[(size_t)(kb + kh + i) * ld + ncol];
        f[i + 8] = base[(size_t)(kb + 16 + kh + i) * ld + ncol];
    }
    return f;
}

__device__ inline float gelu_tanh(float x) {
    float x3 = x * x * x;
    return 0.5f * x * (1.0f + tanhf(0.7978845608028654f * (x + 0.044715f * x3)));
}

// 32-bit LDS byte offset of a __shared__ object (generic -> addrspace(3) cast)
__device__ inline unsigned lds_addr_of(const void* p) {
    return (unsigned)(uintptr_t)(const __attribute__((address_space(3))) void*)p;
}

// Async copy 16B global -> LDS (CDNA5 ASYNCcnt path)
__device__ inline void async_copy_b128(unsigned lds_off, const void* gptr) {
    unsigned long long ga = (unsigned long long)(uintptr_t)gptr;
    asm volatile("global_load_async_to_lds_b128 %0, %1, off"
                 :: "v"(lds_off), "v"(ga) : "memory");
}
__device__ inline void wait_asynccnt0() {
    asm volatile("s_wait_asynccnt 0x0" ::: "memory");
}

// ---------------- tiny conversion kernels ----------------

__global__ void f32_to_bf16_kernel(const float* __restrict__ src, bf16* __restrict__ dst, int n) {
    int i = blockIdx.x * blockDim.x + threadIdx.x;
    if (i < n) dst[i] = (bf16)src[i];
}

// src f32 [batch][K][Nc] -> dst bf16 [batch][Nc][K]
__global__ void transpose_cvt_kernel(const float* __restrict__ src, bf16* __restrict__ dst,
                                     int K, int Nc, int total) {
    int idx = blockIdx.x * blockDim.x + threadIdx.x;
    if (idx >= total) return;
    int per = K * Nc;
    int bb = idx / per, r = idx % per;
    int k = r / Nc, n = r % Nc;
    dst[(size_t)bb * per + (size_t)n * K + k] = (bf16)src[idx];
}

// ---------------- gated dual-conv (implicit GEMM) ----------------
// One block: 16 pixels along W (fixed b,h) x 128 out channels, 4 waves x (2 n-tiles)
template <int BRANCH>
__global__ void conv_moe_kernel(const bf16* __restrict__ Xb, const float* __restrict__ wg,
                                const bf16* __restrict__ WtA, const bf16* __restrict__ WtB,
                                const float* __restrict__ bA, const float* __restrict__ bB,
                                bf16* __restrict__ moe, int coff) {
    __shared__ bf16 lA[16 * 32];
    __shared__ bf16 lB[16 * 32];
    __shared__ float g0s[16], g1s[16];

    int tid = threadIdx.x, lane = tid & 31, wv = tid >> 5;
    int w0 = blockIdx.x * 16, h = blockIdx.y, b = blockIdx.z;
    int tokenBase = (b * 96 + h) * 96 + w0;

    if (tid < 16) {  // per-pixel softmax gate over 2 experts
        const bf16* xp = Xb + (size_t)(tokenBase + tid) * 384 + coff;
        float l0 = 0.f, l1 = 0.f;
        for (int c = 0; c < 128; ++c) {
            float xv = (float)xp[c];
            l0 += xv * wg[c * 2];
            l1 += xv * wg[c * 2 + 1];
        }
        float mx = fmaxf(l0, l1);
        float e0 = __expf(l0 - mx), e1 = __expf(l1 - mx);
        float s = e0 + e1;
        g0s[tid] = e0 / s;
        g1s[tid] = e1 / s;
    }

    v8f accA0 = {}, accA1 = {}, accB0 = {}, accB1 = {};
    for (int t = 0; t < 9; ++t) {
        int dy, dx;
        if constexpr (BRANCH == 0)      { dy = t / 3 - 1; dx = t % 3 - 1; }
        else if constexpr (BRANCH == 1) { dy = t - 4;     dx = 0;         }
        else                            { dy = 0;         dx = t - 4;     }
        for (int kc = 0; kc < 4; ++kc) {
            __syncthreads();
            // stage shifted 16x32 input tiles: threads 0-63 -> conv A (dil1), 64-127 -> conv B (dil2)
            int half = tid >> 6, lt = tid & 63;
            int m = lt >> 2, c0 = (lt & 3) * 8;
            int mul = half ? 2 : 1;
            int hs = h + dy * mul;
            int ws = w0 + m + dx * mul;
            bf16* dst = (half ? lB : lA) + m * 32 + c0;
            if (hs >= 0 && hs < 96 && ws >= 0 && ws < 96) {
                const bf16* src = Xb + (size_t)((b * 96 + hs) * 96 + ws) * 384 + coff + kc * 32 + c0;
#pragma unroll
                for (int j = 0; j < 8; ++j) dst[j] = src[j];
            } else {
#pragma unroll
                for (int j = 0; j < 8; ++j) dst[j] = (bf16)0.f;
            }
            __syncthreads();

            v16bf aA = frag_row(lA + (lane & 15) * 32, lane);
            v16bf aB = frag_row(lB + (lane & 15) * 32, lane);
            int n0 = wv * 32;
            v16bf b0 = frag_row(WtA + (size_t)(t * 128 + n0 + (lane & 15)) * 128 + kc * 32, lane);
            accA0 = wmma_bf16(aA, b0, accA0);
            v16bf b1 = frag_row(WtA + (size_t)(t * 128 + n0 + 16 + (lane & 15)) * 128 + kc * 32, lane);
            accA1 = wmma_bf16(aA, b1, accA1);
            v16bf b2 = frag_row(WtB + (size_t)(t * 128 + n0 + (lane & 15)) * 128 + kc * 32, lane);
            accB0 = wmma_bf16(aB, b2, accB0);
            v16bf b3 = frag_row(WtB + (size_t)(t * 128 + n0 + 16 + (lane & 15)) * 128 + kc * 32, lane);
            accB1 = wmma_bf16(aB, b3, accB1);
        }
    }

    int n = lane & 15;
#pragma unroll
    for (int r = 0; r < 8; ++r) {
        int mm = r + ((lane >> 4) << 3);
        int token = tokenBase + mm;
        {
            int col = wv * 32 + n;
            float va = accA0[r] + bA[col];
            float vb = accB0[r] + bB[col];
            moe[(size_t)token * 128 + col] = (bf16)(g0s[mm] * va + g1s[mm] * vb);
        }
        {
            int col = wv * 32 + 16 + n;
            float va = accA1[r] + bA[col];
            float vb = accB1[r] + bB[col];
            moe[(size_t)token * 128 + col] = (bf16)(g0s[mm] * va + g1s[mm] * vb);
        }
    }
}

// ---------------- generic bf16 GEMM with fused epilogues ----------------
// A: [M x K] bf16 row-major, Bt: [Ncols x K] bf16 (B transposed)
// mode 0: outB = A@B + bias ; 1: outB = gelu(A@B+bias)
// mode 2: outF = beta*outF + gate[row]*(A@B+bias)   (f32)
// mode 3: outF = A@B + bias                          (f32)
__global__ void gemm_bf16_kernel(const bf16* __restrict__ A, const bf16* __restrict__ Bt,
                                 const float* __restrict__ bias, int K, int Ncols,
                                 int outStride, int outColOff, int mode,
                                 const float* __restrict__ gates, int gateIdx, float beta,
                                 bf16* __restrict__ outB, float* __restrict__ outF) {
    __shared__ bf16 As[16 * 32];
    int tid = threadIdx.x, lane = tid & 31, wv = tid >> 5;
    int m0 = blockIdx.x * 16;
    int colBase = blockIdx.y * 64 + wv * 16;

    v8f acc = {};
    int kchunks = K >> 5;
    for (int kc = 0; kc < kchunks; ++kc) {
        __syncthreads();
        // async global -> LDS staging of the 16x32 A tile (1 KB, 64 lanes x B128)
        if (tid < 64) {
            int idx = tid * 8;                    // element index in tile
            int row = idx >> 5, col = idx & 31;   // 4 lanes per 64B row
            const bf16* src = A + (size_t)(m0 + row) * K + kc * 32 + col;
            async_copy_b128(lds_addr_of(As + idx), src);
        }
        // prefetch next weight chunk while waiting
        if (kc + 1 < kchunks)
            __builtin_prefetch(Bt + (size_t)(colBase + (lane & 15)) * K + (kc + 1) * 32, 0, 1);
        wait_asynccnt0();
        __syncthreads();
        v16bf a = frag_row(As + (lane & 15) * 32, lane);
        v16bf b = frag_row(Bt + (size_t)(colBase + (lane & 15)) * K + kc * 32, lane);
        acc = wmma_bf16(a, b, acc);
    }

    int n = lane & 15;
    int col = colBase + n;
    float bv = bias ? bias[col] : 0.f;
#pragma unroll
    for (int r = 0; r < 8; ++r) {
        int mm = r + ((lane >> 4) << 3);
        int row = m0 + mm;
        float v = acc[r] + bv;
        if (mode == 1) v = gelu_tanh(v);
        if (mode <= 1) {
            outB[(size_t)row * outStride + outColOff + col] = (bf16)v;
        } else if (mode == 2) {
            float g = gates[row * 3 + gateIdx];
            float prev = (beta != 0.f) ? outF[(size_t)row * outStride + col] : 0.f;
            outF[(size_t)row * outStride + col] = prev + g * v;
        } else {
            outF[(size_t)row * outStride + outColOff + col] = v;
        }
    }
}

// ---------------- attention: one block per (b, h) ----------------
__global__ void attn_kernel(const bf16* __restrict__ qkv, bf16* __restrict__ obuf) {
    __shared__ float Ss[96 * 96];
    __shared__ bf16  Ps[96 * 96];
    int tid = threadIdx.x, lane = tid & 31, wv = tid >> 5;  // 8 waves
    int bh = blockIdx.x;
    int b = bh / 96, h = bh % 96;
    size_t tokenBase = (size_t)(b * 96 + h) * 96;
    const bf16* Q  = qkv + tokenBase * 384;
    const bf16* Kp = Q + 128;
    const bf16* Vp = Q + 256;

    // S = Q * K^T * scale  (6x6 tiles of 16x16, K=128)
    for (int t = wv; t < 36; t += 8) {
        int ti = t / 6, tj = t % 6;
        v8f acc = {};
        for (int kc = 0; kc < 4; ++kc) {
            v16bf a  = frag_row(Q  + (size_t)(ti * 16 + (lane & 15)) * 384 + kc * 32, lane);
            v16bf bb = frag_row(Kp + (size_t)(tj * 16 + (lane & 15)) * 384 + kc * 32, lane);
            acc = wmma_bf16(a, bb, acc);
        }
        int n = lane & 15;
#pragma unroll
        for (int r = 0; r < 8; ++r) {
            int mm = r + ((lane >> 4) << 3);
            Ss[(ti * 16 + mm) * 96 + tj * 16 + n] = acc[r] * SCALE_F;
        }
    }
    __syncthreads();

    // row softmax
    for (int row = tid; row < 96; row += 256) {
        float mx = -1e30f;
        for (int j = 0; j < 96; ++j) mx = fmaxf(mx, Ss[row * 96 + j]);
        float s = 0.f;
        for (int j = 0; j < 96; ++j) {
            float e = __expf(Ss[row * 96 + j] - mx);
            Ss[row * 96 + j] = e;
            s += e;
        }
        float inv = 1.f / s;
        for (int j = 0; j < 96; ++j) Ps[row * 96 + j] = (bf16)(Ss[row * 96 + j] * inv);
    }
    __syncthreads();

    // O = P * V  (6x8 tiles, K=96); fuse the swapaxes(1,2) into the store
    for (int t = wv; t < 48; t += 8) {
        int ti = t / 8, cj = t % 8;
        v8f acc = {};
        for (int kc = 0; kc < 3; ++kc) {
            v16bf a  = frag_row(Ps + (size_t)(ti * 16 + (lane & 15)) * 96 + kc * 32, lane);
            v16bf bb = frag_col_gather(Vp, 384, cj * 16 + (lane & 15), kc * 32, lane);
            acc = wmma_bf16(a, bb, acc);
        }
        int n = lane & 15;
#pragma unroll
        for (int r = 0; r < 8; ++r) {
            int mm = r + ((lane >> 4) << 3);
            int q = ti * 16 + mm;
            size_t dstTok = (size_t)(b * 96 + q) * 96 + h;  // spatial transpose
            obuf[dstTok * 128 + cj * 16 + n] = (bf16)acc[r];
        }
    }
}

// ---------------- top-2 expert gates ----------------
__global__ void gates_kernel(const bf16* __restrict__ xc, const float* __restrict__ wgf,
                             float* __restrict__ gates) {
    int t = blockIdx.x * blockDim.x + threadIdx.x;
    if (t >= N_TOK) return;
    const bf16* xp = xc + (size_t)t * 384;
    float l[3] = {0.f, 0.f, 0.f};
    for (int c = 0; c < 384; ++c) {
        float xv = (float)xp[c];
        l[0] += xv * wgf[c * 3 + 0];
        l[1] += xv * wgf[c * 3 + 1];
        l[2] += xv * wgf[c * 3 + 2];
    }
    int i0 = 0;
    if (l[1] > l[i0]) i0 = 1;
    if (l[2] > l[i0]) i0 = 2;
    int i1 = -1;
    for (int e = 0; e < 3; ++e) {
        if (e == i0) continue;
        if (i1 < 0 || l[e] > l[i1]) i1 = e;
    }
    float e1 = __expf(l[i1] - l[i0]);
    float g0 = 1.f / (1.f + e1), g1 = e1 / (1.f + e1);
    gates[t * 3 + 0] = 0.f; gates[t * 3 + 1] = 0.f; gates[t * 3 + 2] = 0.f;
    gates[t * 3 + i0] = g0;
    gates[t * 3 + i1] = g1;
}

// ---------------- host launcher ----------------
extern "C" void kernel_launch(void* const* d_in, const int* in_sizes, int n_in,
                              void* d_out, int out_size, void* d_ws, size_t ws_size,
                              hipStream_t stream) {
    const int N = N_TOK;
    char* ws = (char*)d_ws;
    size_t off = 0;
    auto alloc = [&](size_t bytes) -> char* {
        char* p = ws + off;
        off = (off + bytes + 255) & ~(size_t)255;
        return p;
    };

    bf16*  Xb    = (bf16*) alloc((size_t)N * 384 * 2);
    bf16*  moe   = (bf16*) alloc((size_t)N * 128 * 2);
    bf16*  qkvb  = (bf16*) alloc((size_t)N * 384 * 2);
    bf16*  obuf  = (bf16*) alloc((size_t)N * 128 * 2);
    bf16*  xc    = (bf16*) alloc((size_t)N * 384 * 2);
    float* gates = (float*)alloc((size_t)N * 3 * 4);
    float* acc   = (float*)alloc((size_t)N * 384 * 4);
    bf16*  hbuf  = (bf16*) alloc((size_t)N * 1536 * 2);
    bf16*  accb  = hbuf;  // reuse: hbuf dead after last expert GEMM2
    const size_t TAP = 9 * 128 * 128;
    bf16* convWT = (bf16*)alloc(6 * TAP * 2);
    bf16* qkvT   = (bf16*)alloc((size_t)3 * 384 * 128 * 2);
    bf16* aprojT = (bf16*)alloc((size_t)3 * 128 * 128 * 2);
    bf16* m1T    = (bf16*)alloc((size_t)3 * 1536 * 384 * 2);
    bf16* m2T    = (bf16*)alloc((size_t)3 * 384 * 1536 * 2);
    bf16* projT  = (bf16*)alloc((size_t)384 * 384 * 2);

    const float* x = (const float*)d_in[0];

    // input + weight conversion (weights transposed to [n][k] bf16)
    f32_to_bf16_kernel<<<(N * 384 + 255) / 256, 256, 0, stream>>>(x, Xb, N * 384);
    for (int i = 0; i < 6; ++i)
        transpose_cvt_kernel<<<(9 * 128 * 128 + 255) / 256, 256, 0, stream>>>(
            (const float*)d_in[1 + 2 * i], convWT + i * TAP, 128, 128, 9 * 128 * 128);
    transpose_cvt_kernel<<<(3 * 128 * 384 + 255) / 256, 256, 0, stream>>>(
        (const float*)d_in[16], qkvT, 128, 384, 3 * 128 * 384);
    transpose_cvt_kernel<<<(3 * 128 * 128 + 255) / 256, 256, 0, stream>>>(
        (const float*)d_in[17], aprojT, 128, 128, 3 * 128 * 128);
    transpose_cvt_kernel<<<(3 * 384 * 1536 + 255) / 256, 256, 0, stream>>>(
        (const float*)d_in[20], m1T, 384, 1536, 3 * 384 * 1536);
    transpose_cvt_kernel<<<(3 * 1536 * 384 + 255) / 256, 256, 0, stream>>>(
        (const float*)d_in[22], m2T, 1536, 384, 3 * 1536 * 384);
    transpose_cvt_kernel<<<(384 * 384 + 255) / 256, 256, 0, stream>>>(
        (const float*)d_in[24], projT, 384, 384, 384 * 384);

    dim3 convGrid(6, 96, 4);
    for (int i = 0; i < 3; ++i) {
        if (i == 0)
            conv_moe_kernel<0><<<convGrid, 128, 0, stream>>>(
                Xb, (const float*)d_in[13], convWT, convWT + TAP,
                (const float*)d_in[2], (const float*)d_in[4], moe, 0);
        else if (i == 1)
            conv_moe_kernel<1><<<convGrid, 128, 0, stream>>>(
                Xb, (const float*)d_in[14], convWT + 2 * TAP, convWT + 3 * TAP,
                (const float*)d_in[6], (const float*)d_in[8], moe, 128);
        else
            conv_moe_kernel<2><<<convGrid, 128, 0, stream>>>(
                Xb, (const float*)d_in[15], convWT + 4 * TAP, convWT + 5 * TAP,
                (const float*)d_in[10], (const float*)d_in[12], moe, 256);

        dim3 g1(N / 16, 384 / 64);  // qkv = moe @ Wqkv[i]
        gemm_bf16_kernel<<<g1, 128, 0, stream>>>(
            moe, qkvT + (size_t)i * 384 * 128, nullptr, 128, 384,
            384, 0, 0, nullptr, 0, 0.f, qkvb, nullptr);

        attn_kernel<<<4 * 96, 256, 0, stream>>>(qkvb, obuf);

        dim3 g2(N / 16, 128 / 64);  // xc[:, i*128:(i+1)*128] = o @ Wproj[i] + b
        gemm_bf16_kernel<<<g2, 128, 0, stream>>>(
            obuf, aprojT + (size_t)i * 128 * 128, (const float*)d_in[18] + i * 128,
            128, 128, 384, i * 128, 0, nullptr, 0, 0.f, xc, nullptr);
    }

    gates_kernel<<<N / 256, 256, 0, stream>>>(xc, (const float*)d_in[19], gates);

    for (int e = 0; e < 3; ++e) {
        dim3 g3(N / 16, 1536 / 64);  // h = gelu(xc @ W1[e] + b1)
        gemm_bf16_kernel<<<g3, 128, 0, stream>>>(
            xc, m1T + (size_t)e * 1536 * 384, (const float*)d_in[21] + e * 1536,
            384, 1536, 1536, 0, 1, nullptr, 0, 0.f, hbuf, nullptr);
        dim3 g4(N / 16, 384 / 64);   // acc = beta*acc + gate_e * (h @ W2[e] + b2)
        gemm_bf16_kernel<<<g4, 128, 0, stream>>>(
            hbuf, m2T + (size_t)e * 384 * 1536, (const float*)d_in[23] + e * 384,
            1536, 384, 384, 0, 2, gates, e, (e == 0) ? 0.f : 1.f, nullptr, acc);
    }

    f32_to_bf16_kernel<<<(N * 384 + 255) / 256, 256, 0, stream>>>(acc, accb, N * 384);

    dim3 g5(N / 16, 384 / 64);       // out = acc @ Wproj + b (f32 out)
    gemm_bf16_kernel<<<g5, 128, 0, stream>>>(
        accb, projT, (const float*)d_in[25], 384, 384,
        384, 0, 3, nullptr, 0, 0.f, nullptr, (float*)d_out);
}
